// PointExtractor_13924283974189
// MI455X (gfx1250) — matrix operand order repaired and checked by reading, not compiled, not run on hardware
//
#include <hip/hip_runtime.h>
#include <hip/hip_bf16.h>

// ---------------------------------------------------------------------------
// CDNA5 (gfx1250) sparse SubMConv network via v_wmma_f32_16x16x32_bf16.
// Each wave: 32 voxels x 32 output channels (2 A-frags x 2 B-frags = 4 WMMA/k).
// ---------------------------------------------------------------------------

typedef __attribute__((ext_vector_type(16))) __bf16        v16bf;
typedef __attribute__((ext_vector_type(8)))  float         v8f;
typedef __attribute__((ext_vector_type(4)))  unsigned int  v4u;

#define NPTS  700000
#define NVOX  250000
#define MSP   20000
#define CIO   32
#define KVOL  27
#define FRAG_HALFS (KVOL * 2 * 32 * 16)   // per-layer B-fragment halfs = 27648
#define BN_EPS 1e-4f

__device__ __forceinline__ unsigned short f2bf(float f) {
    unsigned int u = __builtin_bit_cast(unsigned int, f);
    unsigned int r = u + 0x7FFFu + ((u >> 16) & 1u);   // round-to-nearest-even
    return (unsigned short)(r >> 16);
}

// ---------------------------------------------------------------------------
// generic zero
// ---------------------------------------------------------------------------
__global__ void k_zero(float* __restrict__ p, int n) {
    int i = blockIdx.x * blockDim.x + threadIdx.x;
    if (i < n) p[i] = 0.0f;
}

// ---------------------------------------------------------------------------
// voxelization scatter: per point, accumulate 6 feature channels + count
// ---------------------------------------------------------------------------
__global__ void k_scatter_vox(const float* __restrict__ feats,
                              const int*   __restrict__ p2v,
                              float* __restrict__ vsum,   // [NVOX,6]
                              float* __restrict__ vcnt,   // [NVOX]
                              int npts) {
    int p = blockIdx.x * blockDim.x + threadIdx.x;
    if (p >= npts) return;
    int v = p2v[p];
    atomicAdd(&vcnt[v], 1.0f);
#pragma unroll
    for (int c = 0; c < 6; ++c)
        atomicAdd(&vsum[v * 6 + c], feats[p * 6 + c]);
}

// ---------------------------------------------------------------------------
// voxel mean -> bf16, channel-padded 6 -> 32
// ---------------------------------------------------------------------------
__global__ void k_vox_to_bf16(const float* __restrict__ vsum,
                              const float* __restrict__ vcnt,
                              unsigned short* __restrict__ abuf, // [NVOX,32] bf16
                              int n32) {
    int i = blockIdx.x * blockDim.x + threadIdx.x;
    if (i >= n32) return;
    int v = i >> 5, c = i & 31;
    float val = 0.0f;
    if (c < 6) val = vsum[v * 6 + c] / fmaxf(vcnt[v], 1.0f);
    abuf[i] = f2bf(val);
}

// ---------------------------------------------------------------------------
// weight prep: build wave32 B-matrix fragments (32x16 bf16 per (k, n-tile)),
// interleaved by 16-byte qword for conflict-minimal LDS reads:
//   halfs layout: [layer][k][t][q(0..1)][lane(0..31)][hh(0..7)]
//   lane half index h = q*8+hh ; K = q*16 + (lane>>4)*8 + hh ; N = t*16+(lane&15)
// layer 0 is the 6->32 input conv, zero-padded to K(channel)=32.
// ---------------------------------------------------------------------------
__global__ void k_prep_wfrag(const float* __restrict__ W_in,     // [27,6,32]
                             const float* __restrict__ W_blocks, // [2,2,27,32,32]
                             unsigned short* __restrict__ wfrag,
                             int total) {
    int i = blockIdx.x * blockDim.x + threadIdx.x;
    if (i >= total) return;
    int hh   = i & 7;
    int lane = (i >> 3) & 31;
    int q    = (i >> 8) & 1;
    int t    = (i >> 9) & 1;
    int k    = (i >> 10) % KVOL;
    int l    = i / FRAG_HALFS;
    int h = q * 8 + hh;
    int K = q * 16 + (lane >> 4) * 8 + hh;
    int N = t * 16 + (lane & 15);
    float w = 0.0f;
    if (l == 0) {
        if (K < 6) w = W_in[(k * 6 + K) * 32 + N];
    } else {
        int b = l - 1; // (rep,sub) flattened
        w = W_blocks[(((size_t)b * KVOL + k) * 32 + K) * 32 + N];
    }
    wfrag[i] = f2bf(w);
}

// ---------------------------------------------------------------------------
// submanifold conv core: one wave = 32 voxels x 32 output channels.
//   D[32x32] = sum_k A_k[32x32(gathered)] * W_k[32x32]  (bf16 WMMA, f32 accum)
// B fragments staged in LDS once per workgroup (54 KB), reused by both A tiles.
// k-loop force-unrolled so gathers are hoisted/clause-batched ahead of WMMAs.
// ---------------------------------------------------------------------------
#define CONV_WAVES 8
#define WAVE_M 32
#define SB_QWORDS (KVOL * 2 * 32 * 2)   // 3456 x 16B = 55296 B

__global__ __launch_bounds__(256) void k_subm_conv_wmma(
        const unsigned short* __restrict__ a,      // [NVOX,32] bf16
        const unsigned short* __restrict__ wfrag,  // this layer's fragments
        const int*   __restrict__ nbr,             // [NVOX,27]
        const float* __restrict__ resid,           // nullptr or [NVOX,32]
        float*       __restrict__ out,             // [NVOX,32]
        int nvox) {
    __shared__ v4u sB[SB_QWORDS];
    const v4u* gB = reinterpret_cast<const v4u*>(wfrag);
    for (int i = threadIdx.x; i < SB_QWORDS; i += 256) sB[i] = gB[i];
    __syncthreads();

    const int wave = threadIdx.x >> 5;
    const int lane = threadIdx.x & 31;
    const int tile = blockIdx.x * CONV_WAVES + wave;
    const int tileBase = tile * WAVE_M;
    if (tileBase >= nvox) return;                 // wave-uniform exit

    const int row = lane & 15;                    // M row / N column index
    const int hi  = lane >> 4;                    // half-wave selector
    const int r0 = min(tileBase + row,      nvox - 1);
    const int r1 = min(tileBase + 16 + row, nvox - 1);
    const int* nrow0 = nbr + (size_t)r0 * KVOL;
    const int* nrow1 = nbr + (size_t)r1 * KVOL;

    v8f acc00 = {0.f,0.f,0.f,0.f,0.f,0.f,0.f,0.f};
    v8f acc01 = acc00, acc10 = acc00, acc11 = acc00;
    const v4u z = {0u, 0u, 0u, 0u};

#pragma unroll
    for (int k = 0; k < KVOL; ++k) {
        int j0 = nrow0[k];
        int j1 = nrow1[k];
        // ---- A fragments: 16x32 bf16 each, ISA lane layout ----
        v16bf af0, af1;
        v4u* a0q = reinterpret_cast<v4u*>(&af0);
        v4u* a1q = reinterpret_cast<v4u*>(&af1);
        if (j0 >= 0) {
            const v4u* p = reinterpret_cast<const v4u*>(a + (size_t)j0 * CIO);
            a0q[0] = p[hi];        // lane<16: K0..7   | lane>=16: K8..15
            a0q[1] = p[hi + 2];    // lane<16: K16..23 | lane>=16: K24..31
        } else { a0q[0] = z; a0q[1] = z; }
        if (j1 >= 0) {
            const v4u* p = reinterpret_cast<const v4u*>(a + (size_t)j1 * CIO);
            a1q[0] = p[hi];
            a1q[1] = p[hi + 2];
        } else { a1q[0] = z; a1q[1] = z; }
        // ---- B fragments from LDS (qword-interleaved) ----
        v16bf b0, b1;
        v4u* b0q = reinterpret_cast<v4u*>(&b0);
        v4u* b1q = reinterpret_cast<v4u*>(&b1);
        int bq0 = (k * 2 + 0) * 64;
        int bq1 = (k * 2 + 1) * 64;
        b0q[0] = sB[bq0 + lane]; b0q[1] = sB[bq0 + 32 + lane];
        b1q[0] = sB[bq1 + lane]; b1q[1] = sB[bq1 + 32 + lane];
        // ---- 4 WMMAs: both M-tiles x both N-tiles ----
        acc00 = __builtin_amdgcn_wmma_f32_16x16x32_bf16(
                    false, af0, false, b0, (short)0, acc00, false, false);
        acc01 = __builtin_amdgcn_wmma_f32_16x16x32_bf16(
                    false, af0, false, b1, (short)0, acc01, false, false);
        acc10 = __builtin_amdgcn_wmma_f32_16x16x32_bf16(
                    false, af1, false, b0, (short)0, acc10, false, false);
        acc11 = __builtin_amdgcn_wmma_f32_16x16x32_bf16(
                    false, af1, false, b1, (short)0, acc11, false, false);
    }

    // D layout: VGPR j -> M = j (lanes 0-15) / j+8 (lanes 16-31), N = lane&15
    const int rowOff = hi * 8;
#pragma unroll
    for (int jj = 0; jj < 8; ++jj) {
        int m0 = tileBase + rowOff + jj;
        if (m0 < nvox) {
            size_t r = (size_t)m0 * CIO;
            float v0 = acc00[jj], v1 = acc01[jj];
            if (resid) { v0 += resid[r + row]; v1 += resid[r + 16 + row]; }
            out[r + row]      = v0;
            out[r + 16 + row] = v1;
        }
        int m1 = tileBase + 16 + rowOff + jj;
        if (m1 < nvox) {
            size_t r = (size_t)m1 * CIO;
            float v0 = acc10[jj], v1 = acc11[jj];
            if (resid) { v0 += resid[r + row]; v1 += resid[r + 16 + row]; }
            out[r + row]      = v0;
            out[r + 16 + row] = v1;
        }
    }
}

// ---------------------------------------------------------------------------
// eval-BN + ReLU; writes bf16 (conv input) or f32 (final activation)
// ---------------------------------------------------------------------------
__global__ void k_bn_relu(const float* __restrict__ x,
                          const float* __restrict__ gamma,
                          const float* __restrict__ beta,
                          unsigned short* __restrict__ out_bf,
                          float* __restrict__ out_f,
                          int n32) {
    int i = blockIdx.x * blockDim.x + threadIdx.x;
    if (i >= n32) return;
    int c = i & 31;
    float sc = gamma[c] * rsqrtf(1.0f + BN_EPS);
    float v  = fmaxf(x[i] * sc + beta[c], 0.0f);
    if (out_bf) out_bf[i] = f2bf(v);
    else        out_f[i]  = v;
}

// ---------------------------------------------------------------------------
// superpoint scatter-mean
// ---------------------------------------------------------------------------
__global__ void k_scatter_sp(const float* __restrict__ xo,   // [NVOX,32]
                             const int*   __restrict__ p2v,
                             const int*   __restrict__ sp,
                             float* __restrict__ ssum,       // [MSP,32]
                             float* __restrict__ scnt,       // [MSP]
                             int n32) {
    int i = blockIdx.x * blockDim.x + threadIdx.x;
    if (i >= n32) return;
    int p = i >> 5, c = i & 31;
    int v = p2v[p], s = sp[p];
    atomicAdd(&ssum[(size_t)s * 32 + c], xo[(size_t)v * 32 + c]);
    if (c == 0) atomicAdd(&scnt[s], 1.0f);
}

__global__ void k_sp_finalize(const float* __restrict__ ssum,
                              const float* __restrict__ scnt,
                              float* __restrict__ out, int n32) {
    int i = blockIdx.x * blockDim.x + threadIdx.x;
    if (i >= n32) return;
    out[i] = ssum[i] / fmaxf(scnt[i >> 5], 1.0f);
}

// ---------------------------------------------------------------------------
// launch
// ---------------------------------------------------------------------------
extern "C" void kernel_launch(void* const* d_in, const int* in_sizes, int n_in,
                              void* d_out, int out_size, void* d_ws, size_t ws_size,
                              hipStream_t stream) {
    (void)in_sizes; (void)n_in; (void)out_size; (void)ws_size;
    const float* feats    = (const float*)d_in[0];
    const float* W_in     = (const float*)d_in[1];
    const float* W_blocks = (const float*)d_in[2];
    const float* bn_gamma = (const float*)d_in[3];
    const float* bn_beta  = (const float*)d_in[4];
    const int*   p2v      = (const int*)d_in[5];
    const int*   nbr      = (const int*)d_in[6];
    const int*   sp       = (const int*)d_in[7];
    float* out = (float*)d_out;

    // ---- carve workspace ----
    char*  ws  = (char*)d_ws;
    size_t off = 0;
    auto carve = [&](size_t bytes) -> void* {
        void* p = ws + off;
        off = (off + bytes + 255) & ~(size_t)255;
        return p;
    };
    float*          vcnt  = (float*)carve((size_t)NVOX * 4);
    float*          vsum  = (float*)carve((size_t)NVOX * 6 * 4);
    float*          xbuf  = (float*)carve((size_t)NVOX * CIO * 4);
    float*          ybuf  = (float*)carve((size_t)NVOX * CIO * 4);
    unsigned short* abuf  = (unsigned short*)carve((size_t)NVOX * CIO * 2);
    unsigned short* wfrag = (unsigned short*)carve((size_t)5 * FRAG_HALFS * 2);
    float*          ssum  = (float*)carve((size_t)MSP * 32 * 4);
    float*          scnt  = (float*)carve((size_t)MSP * 4);

    const int B = 256;
    auto blk = [&](int n) { return (n + B - 1) / B; };

    // 1. zero accumulators (workspace is poisoned)
    k_zero<<<blk(NVOX),     B, 0, stream>>>(vcnt, NVOX);
    k_zero<<<blk(NVOX * 6), B, 0, stream>>>(vsum, NVOX * 6);
    k_zero<<<blk(MSP * 32), B, 0, stream>>>(ssum, MSP * 32);
    k_zero<<<blk(MSP),      B, 0, stream>>>(scnt, MSP);

    // 2. voxelize (segment mean)
    k_scatter_vox<<<blk(NPTS), B, 0, stream>>>(feats, p2v, vsum, vcnt, NPTS);
    k_vox_to_bf16<<<blk(NVOX * 32), B, 0, stream>>>(vsum, vcnt, abuf, NVOX * 32);

    // 3. weight fragment prep (all 5 layers)
    k_prep_wfrag<<<blk(5 * FRAG_HALFS), B, 0, stream>>>(W_in, W_blocks, wfrag,
                                                        5 * FRAG_HALFS);

    // 4. conv stack
    const int nTiles   = (NVOX + WAVE_M - 1) / WAVE_M;
    const int convGrid = (nTiles + CONV_WAVES - 1) / CONV_WAVES;
    const int n32      = NVOX * 32;

    // input conv -> x
    k_subm_conv_wmma<<<convGrid, B, 0, stream>>>(abuf, wfrag + 0 * FRAG_HALFS,
                                                 nbr, nullptr, xbuf, NVOX);
    // block 0
    k_bn_relu<<<blk(n32), B, 0, stream>>>(xbuf, bn_gamma + 0,  bn_beta + 0,  abuf, nullptr, n32);
    k_subm_conv_wmma<<<convGrid, B, 0, stream>>>(abuf, wfrag + 1 * FRAG_HALFS,
                                                 nbr, nullptr, ybuf, NVOX);
    k_bn_relu<<<blk(n32), B, 0, stream>>>(ybuf, bn_gamma + 32, bn_beta + 32, abuf, nullptr, n32);
    k_subm_conv_wmma<<<convGrid, B, 0, stream>>>(abuf, wfrag + 2 * FRAG_HALFS,
                                                 nbr, xbuf, xbuf, NVOX);
    // block 1
    k_bn_relu<<<blk(n32), B, 0, stream>>>(xbuf, bn_gamma + 64, bn_beta + 64, abuf, nullptr, n32);
    k_subm_conv_wmma<<<convGrid, B, 0, stream>>>(abuf, wfrag + 3 * FRAG_HALFS,
                                                 nbr, nullptr, ybuf, NVOX);
    k_bn_relu<<<blk(n32), B, 0, stream>>>(ybuf, bn_gamma + 96, bn_beta + 96, abuf, nullptr, n32);
    k_subm_conv_wmma<<<convGrid, B, 0, stream>>>(abuf, wfrag + 4 * FRAG_HALFS,
                                                 nbr, xbuf, xbuf, NVOX);

    // 5. output BN+ReLU (f32) -> ybuf
    k_bn_relu<<<blk(n32), B, 0, stream>>>(xbuf, bn_gamma + 128, bn_beta + 128,
                                          nullptr, ybuf, n32);

    // 6. devoxelize + superpoint mean
    k_scatter_sp<<<blk(NPTS * 32), B, 0, stream>>>(ybuf, p2v, sp, ssum, scnt, NPTS * 32);
    k_sp_finalize<<<blk(MSP * 32), B, 0, stream>>>(ssum, scnt, out, MSP * 32);
}
